// DCN_layer_illu_1881195676392
// MI455X (gfx1250) — matrix operand order repaired and checked by reading, not compiled, not run on hardware
//
#include <hip/hip_runtime.h>
#include <hip/hip_bf16.h>
#include <math.h>

#define B_    32
#define C_    256
#define O_    256
#define H_    16
#define W_    44
#define HW_   (H_*W_)          // 704
#define KK_   9
#define KDIM_ (C_*KK_)         // 2304
#define MPOS_ 64               // M-tile (positions per block); 704 = 11 * 64
#define MT_   11               // M tiles per batch
#define CG_   32               // channels per staged group
#define KG_   (CG_*KK_)        // 288 = K per staged group (9 chunks of 32)

typedef __attribute__((ext_vector_type(16))) __bf16 v16bf;
typedef __attribute__((ext_vector_type(8)))  __bf16 v8bf;
typedef __attribute__((ext_vector_type(8)))  float  v8f;

static __device__ __forceinline__ unsigned short f2bfu(float f) {
  unsigned u = __builtin_bit_cast(unsigned, f);
  u += 0x7FFFu + ((u >> 16) & 1u);          // round-to-nearest-even
  return (unsigned short)(u >> 16);
}
static __device__ __forceinline__ __bf16 f2bf(float f) {
  unsigned short s = f2bfu(f);
  return __builtin_bit_cast(__bf16, s);
}
static __device__ __forceinline__ int imin(int a, int b) { return a < b ? a : b; }
static __device__ __forceinline__ int imax(int a, int b) { return a > b ? a : b; }

static __device__ __forceinline__ v16bf pack16(v8bf lo, v8bf hi) {
  v16bf r;
#pragma unroll
  for (int q = 0; q < 8; ++q) { r[q] = lo[q]; r[q + 8] = hi[q]; }
  return r;
}

// ---------------- w_off f32 -> bf16, padded to 32 output channels ([32][2304])
__global__ void __launch_bounds__(256) k_wocvt(const float* __restrict__ w_off,
                                               unsigned short* __restrict__ wofb) {
  int i = blockIdx.x * 256 + threadIdx.x;
  if (i >= 32 * KDIM_) return;
  int ch = i / KDIM_;
  wofb[i] = (ch < 27) ? f2bfu(w_off[i]) : (unsigned short)0;
}

// ---------------- offset conv as WMMA GEMM: om[b][27][h][w]
// block = (b, 64-position tile); 8 waves each own one (mt, nt) tile of the
// 4 M-subtiles x 2 N-tiles (N padded 27->32).
__global__ void __launch_bounds__(256) k_offconv_wmma(const float* __restrict__ inter,
                                                      const unsigned short* __restrict__ wofb_u,
                                                      const float* __restrict__ b_off,
                                                      float* __restrict__ om) {
  __shared__ __align__(16) __bf16 As[MPOS_ * KG_];   // 36,864 B
  const __bf16* wofb = (const __bf16*)wofb_u;
  const int tid  = threadIdx.x;
  const int lane = tid & 31;
  const int wid  = tid >> 5;
  const int b    = blockIdx.x / MT_;
  const int pos0 = (blockIdx.x % MT_) * MPOS_;
  const int mt   = wid & 3;          // M-subtile 0..3
  const int nt   = wid >> 2;         // N-tile 0..1
  const int ncol = lane & 15;
  const int hi   = lane >> 4;
  v8f acc = {};

  for (int cg = 0; cg < 8; ++cg) {
    const int c0 = cg * CG_;
    // fill im2col panel: As[pos][ (c-c0)*9 + kk ], lane = channel in group
    for (int idx = wid; idx < MPOS_ * KK_; idx += 8) {
      int pos = idx / KK_, kk = idx % KK_;
      int p = pos0 + pos;
      int h = p / W_, w = p % W_;
      int yy = h - 1 + kk / 3;
      int xx = w - 1 + kk % 3;
      bool valid = (yy >= 0) && (yy < H_) && (xx >= 0) && (xx < W_);
      float v = valid ? inter[(size_t)(b * C_ + c0 + lane) * HW_ + yy * W_ + xx] : 0.0f;
      As[pos * KG_ + lane * KK_ + kk] = f2bf(v);
    }
    __syncthreads();

    const __bf16* wb   = wofb + (size_t)(nt * 16 + ncol) * KDIM_ + cg * KG_ + hi * 16;
    const __bf16* arow = As + (mt * 16 + ncol) * KG_ + hi * 16;
    if (cg < 7) __builtin_prefetch(wb + KG_, 0, 1);   // next group's weights
#pragma unroll 3
    for (int kc = 0; kc < 9; ++kc) {
      int k0 = kc * 32;
      v16bf a  = pack16(*(const v8bf*)(arow + k0), *(const v8bf*)(arow + k0 + 8));
      v16bf bm = pack16(*(const v8bf*)(wb + k0),   *(const v8bf*)(wb + k0 + 8));
      acc = __builtin_amdgcn_wmma_f32_16x16x32_bf16(false, a, false, bm,
                                                    (short)0, acc, false, false);
    }
    __syncthreads();
  }

  int o = nt * 16 + ncol;
  if (o < 27) {
    float bs = b_off[o];
#pragma unroll
    for (int j = 0; j < 8; ++j) {
      int p = pos0 + mt * 16 + j + hi * 8;
      om[(size_t)(b * 27 + o) * HW_ + p] = acc[j] + bs;
    }
  }
}

// ---------------- illumination: resize(align_corners) + inv + global min/max
__global__ void k_minmax_init(unsigned* mm) {
  if (threadIdx.x == 0) { mm[0] = 0x7F800000u; mm[1] = 0u; }
}

__global__ void __launch_bounds__(256) k_illu_inv(const float* __restrict__ illu,
                                                  float* __restrict__ inv,
                                                  unsigned* __restrict__ mm) {
  int i = blockIdx.x * 256 + threadIdx.x;
  if (i >= B_ * HW_) return;
  int p = i % HW_, b = i / HW_;
  int h = p / W_, w = p % W_;
  float ys = (63.0f / 15.0f) * (float)h;     // linspace(0, 63, 16)
  float xs = (175.0f / 43.0f) * (float)w;    // linspace(0, 175, 44)
  float y0f = floorf(ys), x0f = floorf(xs);
  int y0 = (int)y0f, x0 = (int)x0f;
  int y1 = imin(y0 + 1, 63), x1 = imin(x0 + 1, 175);
  float wy = ys - y0f, wx = xs - x0f;
  const float* im = illu + (size_t)b * 64 * 176;
  float v = im[y0 * 176 + x0] * (1.0f - wy) * (1.0f - wx)
          + im[y0 * 176 + x1] * (1.0f - wy) * wx
          + im[y1 * 176 + x0] * wy * (1.0f - wx)
          + im[y1 * 176 + x1] * wy * wx;
  float r = 1.0f / (v + 1e-6f);
  inv[i] = r;
  atomicMin(&mm[0], __float_as_uint(r));   // positive floats: uint order == float order
  atomicMax(&mm[1], __float_as_uint(r));
}

__global__ void __launch_bounds__(256) k_scale(const float* __restrict__ inv,
                                               const unsigned* __restrict__ mm,
                                               float* __restrict__ scl) {
  int i = blockIdx.x * 256 + threadIdx.x;
  if (i >= B_ * HW_) return;
  float mn = __uint_as_float(mm[0]);
  float mx = __uint_as_float(mm[1]);
  scl[i] = 1.0f + 0.5f * (inv[i] - mn) / (mx - mn);
}

// ---------------- weight f32 -> bf16 (B panel: [O][K=2304], K contiguous)
__global__ void __launch_bounds__(256) k_wcvt(const float* __restrict__ w,
                                              unsigned short* __restrict__ wb) {
  int i = blockIdx.x * 256 + threadIdx.x;
  if (i >= O_ * KDIM_) return;
  wb[i] = f2bfu(w[i]);
}

// ---------------- fused deformable sample + bf16 WMMA GEMM
// block = (b, 64-position tile). 8 waves; each wave owns 2 N-tiles x 4 M-subtiles.
__global__ void __launch_bounds__(256) k_dcn_wmma(const float* __restrict__ feat,
                                                  const float* __restrict__ om,
                                                  const float* __restrict__ scl,
                                                  const unsigned short* __restrict__ wbf_u,
                                                  const float* __restrict__ bias,
                                                  float* __restrict__ out) {
  __shared__ __align__(16) __bf16 As[MPOS_ * KG_];   // 36,864 B
  __shared__ int   s_lin[MPOS_ * KK_][4];            //  9,216 B
  __shared__ float s_wgt[MPOS_ * KK_][4];            //  9,216 B

  const __bf16* wbf = (const __bf16*)wbf_u;
  const int tid  = threadIdx.x;
  const int lane = tid & 31;
  const int wid  = tid >> 5;
  const int b    = blockIdx.x / MT_;
  const int pos0 = (blockIdx.x % MT_) * MPOS_;
  const int ncol = lane & 15;
  const int hi   = lane >> 4;

  // --- phase 1: bilinear corner table per (pos, tap), mask & validity folded in
  for (int t = tid; t < MPOS_ * KK_; t += 256) {
    int pos = t / KK_, kk = t % KK_;
    int p = pos0 + pos;
    int h = p / W_, w = p % W_;
    const float* omb = om + (size_t)b * 27 * HW_;
    float sc = scl[b * HW_ + p];
    float dy = omb[(2 * kk) * HW_ + p] * sc;
    float dx = omb[(2 * kk + 1) * HW_ + p] * sc;
    float mk = omb[(18 + kk) * HW_ + p];
    mk = 1.0f / (1.0f + __expf(-mk));
    float py = (float)(h - 1 + kk / 3) + dy;
    float px = (float)(w - 1 + kk % 3) + dx;
    float y0f = floorf(py), x0f = floorf(px);
    int y0 = (int)y0f, x0 = (int)x0f;
    float wy = py - y0f, wx = px - x0f;
#pragma unroll
    for (int c4 = 0; c4 < 4; ++c4) {
      int yy = y0 + (c4 >> 1);
      int xx = x0 + (c4 & 1);
      float wgt = ((c4 >> 1) ? wy : (1.0f - wy)) * ((c4 & 1) ? wx : (1.0f - wx)) * mk;
      bool valid = (yy >= 0) && (yy < H_) && (xx >= 0) && (xx < W_);
      int yc = imin(imax(yy, 0), H_ - 1);
      int xc = imin(imax(xx, 0), W_ - 1);
      s_lin[t][c4] = yc * W_ + xc;
      s_wgt[t][c4] = valid ? wgt : 0.0f;
    }
  }
  __syncthreads();

  const int nt0 = wid * 2, nt1 = nt0 + 1;
  v8f acc0[4] = {{}, {}, {}, {}};
  v8f acc1[4] = {{}, {}, {}, {}};

  for (int cg = 0; cg < 8; ++cg) {
    const int c0 = cg * CG_;
    // --- fill bf16 A panel: As[pos][ (c-c0)*9 + kk ], lane = channel in group
    for (int idx = wid; idx < MPOS_ * KK_; idx += 8) {
      int pos = idx / KK_, kk = idx % KK_;
      int   l0 = s_lin[idx][0], l1 = s_lin[idx][1], l2 = s_lin[idx][2], l3 = s_lin[idx][3];
      float w0 = s_wgt[idx][0], w1 = s_wgt[idx][1], w2 = s_wgt[idx][2], w3 = s_wgt[idx][3];
      const float* fb = feat + (size_t)(b * C_ + c0 + lane) * HW_;
      float v = w0 * fb[l0] + w1 * fb[l1] + w2 * fb[l2] + w3 * fb[l3];
      As[pos * KG_ + lane * KK_ + kk] = f2bf(v);
    }
    __syncthreads();

    // --- GEMM: 9 k-chunks of 32 over this channel group
    const __bf16* wb0 = wbf + (size_t)(nt0 * 16 + ncol) * KDIM_ + cg * KG_ + hi * 16;
    const __bf16* wb1 = wbf + (size_t)(nt1 * 16 + ncol) * KDIM_ + cg * KG_ + hi * 16;
    if (cg < 7) {   // prefetch next group's weight region (global_prefetch_b8)
      __builtin_prefetch(wb0 + KG_, 0, 1);
      __builtin_prefetch(wb1 + KG_, 0, 1);
    }
#pragma unroll 3
    for (int kc = 0; kc < 9; ++kc) {
      int k0 = kc * 32;
      // B frags: lane = N column; lanes 0-15 K k0..k0+15, lanes 16-31 K k0+16..k0+31
      v16bf bm0 = pack16(*(const v8bf*)(wb0 + k0), *(const v8bf*)(wb0 + k0 + 8));
      v16bf bm1 = pack16(*(const v8bf*)(wb1 + k0), *(const v8bf*)(wb1 + k0 + 8));
#pragma unroll
      for (int mt = 0; mt < 4; ++mt) {
        // A frag: lanes 0-15 K 0..15; lanes 16-31 K 16..31
        const __bf16* arow = As + (mt * 16 + ncol) * KG_ + hi * 16;
        v16bf a = pack16(*(const v8bf*)(arow + k0), *(const v8bf*)(arow + k0 + 8));
        acc0[mt] = __builtin_amdgcn_wmma_f32_16x16x32_bf16(false, a, false, bm0,
                                                           (short)0, acc0[mt], false, false);
        acc1[mt] = __builtin_amdgcn_wmma_f32_16x16x32_bf16(false, a, false, bm1,
                                                           (short)0, acc1[mt], false, false);
      }
    }
    __syncthreads();
  }

  // --- store D: lane -> N = ncol; VGPR j -> M = j + hi*8
  int o0 = nt0 * 16 + ncol;
  int o1 = nt1 * 16 + ncol;
  float bs0 = bias[o0], bs1 = bias[o1];
#pragma unroll
  for (int mt = 0; mt < 4; ++mt) {
#pragma unroll
    for (int j = 0; j < 8; ++j) {
      int p = pos0 + mt * 16 + j + hi * 8;
      out[(size_t)(b * O_ + o0) * HW_ + p] = acc0[mt][j] + bs0;
      out[(size_t)(b * O_ + o1) * HW_ + p] = acc1[mt][j] + bs1;
    }
  }
}

extern "C" void kernel_launch(void* const* d_in, const int* in_sizes, int n_in,
                              void* d_out, int out_size, void* d_ws, size_t ws_size,
                              hipStream_t stream) {
  const float* feat   = (const float*)d_in[0];   // input_feat (32,256,16,44)
  const float* inter  = (const float*)d_in[1];   // inter      (32,256,16,44)
  const float* illu   = (const float*)d_in[2];   // illu_map   (32,1,64,176)
  const float* weight = (const float*)d_in[3];   // (256,256,3,3)
  const float* bias   = (const float*)d_in[4];   // (256,)
  const float* w_off  = (const float*)d_in[5];   // (27,256,3,3)
  const float* b_off  = (const float*)d_in[6];   // (27,)
  float* out = (float*)d_out;

  char* ws = (char*)d_ws;
  float*          om   = (float*)(ws);                    // 32*27*704*4   = 2,433,024 B
  float*          inv  = (float*)(ws + 2433024);          // 22528*4       =    90,112 B
  unsigned*       mm   = (unsigned*)(ws + 2523136);       //                       16 B
  float*          scl  = (float*)(ws + 2523152);          // 22528*4       =    90,112 B
  unsigned short* wbf  = (unsigned short*)(ws + 2613264); // 589824*2      = 1,179,648 B
  unsigned short* wofb = (unsigned short*)(ws + 3792912); // 32*2304*2     =   147,456 B

  k_wocvt<<<(32 * KDIM_ + 255) / 256, 256, 0, stream>>>(w_off, wofb);
  k_offconv_wmma<<<B_ * MT_, 256, 0, stream>>>(inter, wofb, b_off, om);
  k_minmax_init<<<1, 32, 0, stream>>>(mm);
  k_illu_inv<<<(B_ * HW_ + 255) / 256, 256, 0, stream>>>(illu, inv, mm);
  k_scale<<<(B_ * HW_ + 255) / 256, 256, 0, stream>>>(inv, mm, scl);
  k_wcvt<<<(O_ * KDIM_ + 255) / 256, 256, 0, stream>>>(weight, wbf);
  k_dcn_wmma<<<B_ * MT_, 256, 0, stream>>>(feat, om, scl, wbf, bias, out);
}
